// PointTransformerNet_28879360098871
// MI455X (gfx1250) — compile-verified, matrix operand order checked by reading, not run on hardware
//
#include <hip/hip_runtime.h>
#include <hip/hip_bf16.h>
#include <math.h>

typedef __attribute__((ext_vector_type(16))) _Float16 v16h;
typedef __attribute__((ext_vector_type(8)))  float    v8f;

#define BB 16
#define NN 2048
#define KK 5
#define RTOT (BB * NN)     // 32768 rows (all graphs flattened)
#define EG (NN * 6)        // 12288 edges per graph (5 knn + self)

// ---------------------------------------------------------------------------
// Weight pre-transpose + f32->f16 convert: Wt[n*K + k] = (f16) W[k*N + n].
// Makes the WMMA B-fragment loads contiguous (same element->K map as A).
// ---------------------------------------------------------------------------
__global__ void wtrans_f16_kernel(const float* __restrict__ W, _Float16* __restrict__ Wt,
                                  int K, int N)
{
    int t = blockIdx.x * blockDim.x + threadIdx.x;
    if (t >= K * N) return;
    int k = t / N, n = t % N;                 // coalesced read of W
    Wt[(size_t)n * K + k] = (_Float16)W[t];
}

// ---------------------------------------------------------------------------
// WMMA GEMM: C[M,N] = act(A[M,K] @ W[K,N] + bias), W given as f16 Wt[N,K].
// One wave computes an (MT*16)x16 strip: MT M-tiles share one B fragment per
// k-step (MT back-to-back v_wmma per B load, no row guards -> straight-line).
// Requirements: K % 32 == 0, M % (MT*16) == 0. N arbitrary (column-guarded).
// ---------------------------------------------------------------------------
template <int MT>
__global__ __launch_bounds__(32) void gemm16_wmma(
    const float* __restrict__ A, const _Float16* __restrict__ Wt,
    const float* __restrict__ bias, float* __restrict__ C,
    int M, int K, int N, int relu)
{
    const int lane = threadIdx.x & 31;
    const int half = lane >> 4;               // which 16-lane half
    const int l16  = lane & 15;
    const int m0   = blockIdx.x * (MT * 16);  // strip base row
    const int col  = blockIdx.y * 16 + l16;   // B/C column for this lane

    v8f acc[MT];
#pragma unroll
    for (int mt = 0; mt < MT; ++mt) acc[mt] = (v8f){};

    for (int k0 = 0; k0 < K; k0 += 32) {
        // B fragment: element e -> K = k0 + half*8 + e (e<8), +16 for e>=8.
        // Wt row-major [N,K] => both 8-element groups contiguous (16B each).
        v16h bf = {};
        if (col < N) {
            const _Float16* wq = Wt + (size_t)col * K + k0 + half * 8;
#pragma unroll
            for (int e = 0; e < 8; ++e) bf[e] = wq[e];
#pragma unroll
            for (int e = 0; e < 8; ++e) bf[8 + e] = wq[16 + e];
        }
#pragma unroll
        for (int mt = 0; mt < MT; ++mt) {
            v16h af;
            const float* ap = A + (size_t)(m0 + mt * 16 + l16) * K + k0 + half * 8;
            __builtin_prefetch(ap + 32, 0, 1);     // next k-tile
#pragma unroll
            for (int e = 0; e < 8; ++e) af[e] = (_Float16)ap[e];
#pragma unroll
            for (int e = 0; e < 8; ++e) af[8 + e] = (_Float16)ap[16 + e];
            acc[mt] = __builtin_amdgcn_wmma_f32_16x16x32_f16(
                /*neg_a=*/false, af, /*neg_b=*/false, bf,
                /*c_mod=*/(short)0, acc[mt], /*reuse_a=*/false, /*reuse_b=*/false);
        }
    }
    if (col < N) {
        const float bv = bias ? bias[col] : 0.0f;
#pragma unroll
        for (int mt = 0; mt < MT; ++mt) {
#pragma unroll
            for (int e = 0; e < 8; ++e) {
                const int r = m0 + mt * 16 + half * 8 + e;   // D layout: VGPR e -> M
                float v = acc[mt][e] + bv;
                if (relu) v = fmaxf(v, 0.0f);
                C[(size_t)r * N + col] = v;
            }
        }
    }
}

// ---------------------------------------------------------------------------
// Input linear (K = 6, too small for WMMA): y[r,n] = x[r,:6] @ W[6,64] + b
// ---------------------------------------------------------------------------
__global__ void lin_in_kernel(const float* __restrict__ x, const float* __restrict__ W,
                              const float* __restrict__ b, float* __restrict__ y, int R)
{
    int t = blockIdx.x * blockDim.x + threadIdx.x;
    if (t >= R * 64) return;
    int r = t >> 6, n = t & 63;
    float s = b[n];
#pragma unroll
    for (int k = 0; k < 6; ++k) s += x[(size_t)r * 6 + k] * W[k * 64 + n];
    y[t] = s;
}

// ---------------------------------------------------------------------------
// BatchNorm (training mode): per-channel mean/invstd over all R rows.
// ---------------------------------------------------------------------------
__global__ __launch_bounds__(256) void bn_reduce_kernel(
    const float* __restrict__ y, float* __restrict__ stats, int R, int C)
{
    __shared__ float ss[256], sq[256];
    const int c = blockIdx.x, t = threadIdx.x;
    float s = 0.f, q = 0.f;
    for (int r = t; r < R; r += 256) {
        float v = y[(size_t)r * C + c];
        s += v; q += v * v;
    }
    ss[t] = s; sq[t] = q; __syncthreads();
    for (int off = 128; off; off >>= 1) {
        if (t < off) { ss[t] += ss[t + off]; sq[t] += sq[t + off]; }
        __syncthreads();
    }
    if (t == 0) {
        float m = ss[0] / (float)R;
        float var = sq[0] / (float)R - m * m;
        stats[c] = m;
        stats[C + c] = rsqrtf(var + 1e-5f);
    }
}

__global__ void bn_apply_relu_kernel(float* __restrict__ y, const float* __restrict__ stats,
                                     const float* __restrict__ g, const float* __restrict__ be,
                                     int R, int C)
{
    size_t t = (size_t)blockIdx.x * blockDim.x + threadIdx.x;
    if (t >= (size_t)R * C) return;
    int c = (int)(t % C);
    float v = (y[t] - stats[c]) * stats[C + c] * g[c] + be[c];
    y[t] = fmaxf(v, 0.0f);
}

// ---------------------------------------------------------------------------
// pos_nn layer 1 for one graph: ph[e,h] = relu((pos_i - pos_j) @ W1[3,64] + b1)
// edge e = i*6 + s; slot 0 = self-loop, slots 1..5 = knn-graph neighbors.
// ---------------------------------------------------------------------------
__global__ void pos_l1_kernel(const float* __restrict__ pos, const int* __restrict__ nbrg,
                              const float* __restrict__ W1, const float* __restrict__ b1,
                              float* __restrict__ ph, int g)
{
    int t = blockIdx.x * blockDim.x + threadIdx.x;
    if (t >= EG * 64) return;
    int h = t & 63, e = t >> 6, i = e / 6, s = e % 6;
    int j = (s == 0) ? i : nbrg[((size_t)g * NN + i) * KK + (s - 1)];
    const float* pi = pos + ((size_t)g * NN + i) * 3;
    const float* pj = pos + ((size_t)g * NN + j) * 3;
    float d0 = pi[0] - pj[0], d1 = pi[1] - pj[1], d2 = pi[2] - pj[2];
    float v = b1[h] + d0 * W1[0 * 64 + h] + d1 * W1[1 * 64 + h] + d2 * W1[2 * 64 + h];
    ph[t] = fmaxf(v, 0.0f);
}

// t[e,ch] = a_dst[i,ch] - a_src[j,ch] + delta[e,ch]   (one graph)
__global__ void tbuild_kernel(const float* __restrict__ AD, const float* __restrict__ AS,
                              const float* __restrict__ DEL, const int* __restrict__ nbrg,
                              float* __restrict__ T, int g, int C)
{
    int t = blockIdx.x * blockDim.x + threadIdx.x;
    if (t >= EG * C) return;
    int ch = t % C, e = t / C, i = e / 6, s = e % 6;
    int j = (s == 0) ? i : nbrg[((size_t)g * NN + i) * KK + (s - 1)];
    T[t] = AD[((size_t)g * NN + i) * C + ch] - AS[((size_t)g * NN + j) * C + ch] + DEL[t];
}

// softmax over the 6 slots (per target, per channel) + weighted sum of (v_j + delta)
__global__ void aggregate_kernel(const float* __restrict__ AL, const float* __restrict__ DEL,
                                 const float* __restrict__ V, const int* __restrict__ nbrg,
                                 float* __restrict__ OUT, int g, int C)
{
    int t = blockIdx.x * blockDim.x + threadIdx.x;
    if (t >= NN * C) return;
    int ch = t % C, i = t / C;
    int js[6];
    js[0] = i;
#pragma unroll
    for (int s = 1; s < 6; ++s) js[s] = nbrg[((size_t)g * NN + i) * KK + (s - 1)];
    float a[6], m = -1e30f;
#pragma unroll
    for (int s = 0; s < 6; ++s) {
        a[s] = AL[((size_t)i * 6 + s) * C + ch];
        m = fmaxf(m, a[s]);
    }
    float den = 0.f, num = 0.f;
#pragma unroll
    for (int s = 0; s < 6; ++s) {
        float ea = expf(a[s] - m);
        den += ea;
        num += ea * (V[((size_t)g * NN + js[s]) * C + ch] + DEL[((size_t)i * 6 + s) * C + ch]);
    }
    OUT[((size_t)g * NN + i) * C + ch] = num / den;
}

// TransitionDown pooling: x[r,ch] = max over 5 nbr_d rows (self included in knn)
__global__ void maxpool_kernel(const float* __restrict__ Y, const int* __restrict__ nbrd,
                               float* __restrict__ X, int C)
{
    size_t t = (size_t)blockIdx.x * blockDim.x + threadIdx.x;
    if (t >= (size_t)RTOT * C) return;
    int ch = (int)(t % C);
    int r = (int)(t / C);
    int gbase = (r / NN) * NN;
    float m = -1e30f;
#pragma unroll
    for (int s = 0; s < KK; ++s) {
        int j = nbrd[(size_t)r * KK + s];
        m = fmaxf(m, Y[((size_t)(gbase + j)) * C + ch]);
    }
    X[t] = m;
}

// global mean pool: P[g,ch] = mean over N points
__global__ __launch_bounds__(256) void meanpool_kernel(
    const float* __restrict__ Y, float* __restrict__ P, int C)
{
    __shared__ float sm[256];
    int gc = blockIdx.x;                // g*C + ch
    int g = gc / C, ch = gc % C;
    float s = 0.f;
    for (int i = threadIdx.x; i < NN; i += 256)
        s += Y[((size_t)g * NN + i) * C + ch];
    sm[threadIdx.x] = s; __syncthreads();
    for (int off = 128; off; off >>= 1) {
        if (threadIdx.x < off) sm[threadIdx.x] += sm[threadIdx.x + off];
        __syncthreads();
    }
    if (threadIdx.x == 0) P[gc] = sm[0] / (float)NN;
}

// ---------------------------------------------------------------------------
// Host orchestration
// ---------------------------------------------------------------------------
extern "C" void kernel_launch(void* const* d_in, const int* in_sizes, int n_in,
                              void* d_out, int out_size, void* d_ws, size_t ws_size,
                              hipStream_t stream)
{
    (void)in_sizes; (void)n_in; (void)out_size; (void)ws_size;
    auto f = [&](int i) { return (const float*)d_in[i]; };

    const float* x_in = f(0);
    const float* pos  = f(1);
    // params leaves in JAX tree_leaves order (dict keys sorted at every level)
    const float* in_W   = f(2);
    const float* in_b   = f(3);
    const float* in_be  = f(4);
    const float* in_g   = f(5);
    const float* out_W1 = f(6);
    const float* out_W2 = f(7);
    const float* out_W3 = f(8);
    const float* out_b1 = f(9);
    const float* out_b2 = f(10);
    const float* out_b3 = f(11);
    struct TBp {
        const float *Wdst, *Wlin, *Wsrc;
        const float *aW1, *aW2, *ab1, *ab2;
        const float *linInW, *linInb, *linOutW, *linOutb;
        const float *pW1, *pW2, *pb1, *pb2;
    } tbp[3];
    for (int t = 0; t < 3; ++t) {
        int o = 12 + 15 * t;
        tbp[t].Wdst = f(o + 0); tbp[t].Wlin = f(o + 1); tbp[t].Wsrc = f(o + 2);
        tbp[t].aW1 = f(o + 3); tbp[t].aW2 = f(o + 4); tbp[t].ab1 = f(o + 5); tbp[t].ab2 = f(o + 6);
        tbp[t].linInW = f(o + 7); tbp[t].linInb = f(o + 8);
        tbp[t].linOutW = f(o + 9); tbp[t].linOutb = f(o + 10);
        tbp[t].pW1 = f(o + 11); tbp[t].pW2 = f(o + 12); tbp[t].pb1 = f(o + 13); tbp[t].pb2 = f(o + 14);
    }
    const float* tdW[2]  = { f(57), f(61) };
    const float* tdb[2]  = { f(58), f(62) };
    const float* tdbe[2] = { f(59), f(63) };
    const float* tdg[2]  = { f(60), f(64) };
    const int* nbrg = (const int*)d_in[65];
    const int* nbrd = (const int*)d_in[66];

    // workspace carve-up
    char* wp = (char*)d_ws;
    auto alloc = [&](size_t bytes) {
        void* p = (void*)wp;
        wp += (bytes + 255) & ~(size_t)255;
        return p;
    };
    float* XN  = (float*)alloc((size_t)RTOT * 256 * 4);  // node features (block in/out)
    float* H   = (float*)alloc((size_t)RTOT * 256 * 4);  // hidden / TD pre-pool
    float* V   = (float*)alloc((size_t)RTOT * 256 * 4);  // value projection
    float* ASb = (float*)alloc((size_t)RTOT * 256 * 4);  // a_src
    float* ADb = (float*)alloc((size_t)RTOT * 256 * 4);  // a_dst
    float* CV  = (float*)alloc((size_t)RTOT * 256 * 4);  // conv output
    float* PH  = (float*)alloc((size_t)EG * 64 * 4);     // pos_nn hidden (per graph)
    float* DEL = (float*)alloc((size_t)EG * 256 * 4);    // delta (per graph)
    float* TT  = (float*)alloc((size_t)EG * 256 * 4);    // attn input (per graph)
    float* AH  = (float*)alloc((size_t)EG * 64 * 4);     // attn hidden (per graph)
    float* AL  = (float*)alloc((size_t)EG * 256 * 4);    // attn logits (per graph)
    float* STATS = (float*)alloc(512 * 4);               // bn mean | invstd
    float* POOL  = (float*)alloc(16 * 256 * 4);
    float* O1    = (float*)alloc(16 * 64 * 4);
    float* O2    = (float*)alloc(16 * 64 * 4);
    _Float16* WT[8];
    for (int i = 0; i < 8; ++i) WT[i] = (_Float16*)alloc((size_t)256 * 256 * 2);

    auto cdiv = [](int a, int b) { return (a + b - 1) / b; };
    auto xpose = [&](const float* W, _Float16* Wt, int K, int N) {
        wtrans_f16_kernel<<<cdiv(K * N, 256), 256, 0, stream>>>(W, Wt, K, N);
    };
    auto gemm = [&](const float* A, const _Float16* Wt, const float* bias, float* C,
                    int M, int K, int N, int relu) {
        if (M % 64 == 0) {
            dim3 grid(M / 64, cdiv(N, 16));
            gemm16_wmma<4><<<grid, dim3(32), 0, stream>>>(A, Wt, bias, C, M, K, N, relu);
        } else {
            dim3 grid(M / 16, cdiv(N, 16));
            gemm16_wmma<1><<<grid, dim3(32), 0, stream>>>(A, Wt, bias, C, M, K, N, relu);
        }
    };

    auto tblock = [&](const TBp& p, int C) {
        // transpose all weights for this block once (amortized over 16 graphs)
        xpose(p.linInW, WT[0], C, C);
        xpose(p.Wlin,  WT[1], C, C);
        xpose(p.Wsrc,  WT[2], C, C);
        xpose(p.Wdst,  WT[3], C, C);
        xpose(p.pW2,   WT[4], 64, C);
        xpose(p.aW1,   WT[5], C, 64);
        xpose(p.aW2,   WT[6], 64, C);
        xpose(p.linOutW, WT[7], C, C);
        // h = relu(x @ lin_in + b)
        gemm(XN, WT[0], p.linInb, H, RTOT, C, C, 1);
        // projections
        gemm(H, WT[1], nullptr, V,   RTOT, C, C, 0);
        gemm(H, WT[2], nullptr, ASb, RTOT, C, C, 0);
        gemm(H, WT[3], nullptr, ADb, RTOT, C, C, 0);
        // per-graph edge pipeline
        for (int g = 0; g < BB; ++g) {
            pos_l1_kernel<<<cdiv(EG * 64, 256), 256, 0, stream>>>(pos, nbrg, p.pW1, p.pb1, PH, g);
            gemm(PH, WT[4], p.pb2, DEL, EG, 64, C, 1);                       // delta
            tbuild_kernel<<<cdiv(EG * C, 256), 256, 0, stream>>>(ADb, ASb, DEL, nbrg, TT, g, C);
            gemm(TT, WT[5], p.ab1, AH, EG, C, 64, 1);                        // attn hidden
            gemm(AH, WT[6], p.ab2, AL, EG, 64, C, 1);                        // attn logits
            aggregate_kernel<<<cdiv(NN * C, 256), 256, 0, stream>>>(AL, DEL, V, nbrg, CV, g, C);
        }
        // y = relu(conv @ lin_out + b)  -> back into XN
        gemm(CV, WT[7], p.linOutb, XN, RTOT, C, C, 1);
    };

    // ---- input linear (K=6) + BN + ReLU ----
    lin_in_kernel<<<cdiv(RTOT * 64, 256), 256, 0, stream>>>(x_in, in_W, in_b, XN, RTOT);
    bn_reduce_kernel<<<64, 256, 0, stream>>>(XN, STATS, RTOT, 64);
    bn_apply_relu_kernel<<<cdiv(RTOT * 64, 256), 256, 0, stream>>>(XN, STATS, in_g, in_be, RTOT, 64);

    // ---- transformer block @ 64 ----
    tblock(tbp[0], 64);

    // ---- TransitionDown + transformer block @ 128 and @ 256 ----
    const int dims[3] = { 64, 128, 256 };
    for (int i = 0; i < 2; ++i) {
        int Cn = dims[i + 1];
        xpose(tdW[i], WT[0], dims[i], Cn);
        gemm(XN, WT[0], tdb[i], H, RTOT, dims[i], Cn, 0);
        bn_reduce_kernel<<<Cn, 256, 0, stream>>>(H, STATS, RTOT, Cn);
        bn_apply_relu_kernel<<<cdiv(RTOT * Cn, 256), 256, 0, stream>>>(H, STATS, tdg[i], tdbe[i], RTOT, Cn);
        maxpool_kernel<<<cdiv(RTOT * Cn, 256), 256, 0, stream>>>(H, nbrd, XN, Cn);
        tblock(tbp[i + 1], Cn);
    }

    // ---- global mean pool + classifier head ----
    meanpool_kernel<<<BB * 256, 256, 0, stream>>>(XN, POOL, 256);
    xpose(out_W1, WT[0], 256, 64);
    xpose(out_W2, WT[1], 64, 64);
    xpose(out_W3, WT[2], 64, 2);
    gemm(POOL, WT[0], out_b1, O1, 16, 256, 64, 1);
    gemm(O1, WT[1], out_b2, O2, 16, 64, 64, 1);
    gemm(O2, WT[2], out_b3, (float*)d_out, 16, 64, 2, 0);
}
